// Decoder_71614284693889
// MI455X (gfx1250) — matrix operand order repaired and checked by reading, not compiled, not run on hardware
//
#include <hip/hip_runtime.h>
#include <hip/hip_bf16.h>

typedef __attribute__((ext_vector_type(16))) __bf16 v16bf;
typedef __attribute__((ext_vector_type(8)))  __bf16 v8bf;
typedef __attribute__((ext_vector_type(8)))  float  v8f;

#define Bsz  128
#define Tsz  512
#define Hsz  1024
#define Osz  128
#define NWG  64          // persistent workgroups; each owns 16 hidden columns
#define NCOL 16
#define KIT  (Hsz / 32)  // 32 k-iterations of 16x16x32 WMMA

__device__ __forceinline__ float sigmoidf_(float x) {
    return 1.0f / (1.0f + __expf(-x));
}

// ---------------------------------------------------------------------------
// Init: h0 (fp32) -> bf16 h buffer 0; zero the grid-barrier state.
// ---------------------------------------------------------------------------
__global__ __launch_bounds__(256)
void lstm_init(const float* __restrict__ h0, __bf16* __restrict__ hb0,
               unsigned* __restrict__ bar) {
    int i = blockIdx.x * 256 + threadIdx.x;
    if (i < Bsz * Hsz) hb0[i] = (__bf16)h0[i];
    if (i == 0) { bar[0] = 0u; bar[1] = 0u; }
}

// ---------------------------------------------------------------------------
// Persistent LSTM: W_hh slice resident in LDS (pre-swizzled B fragments),
// c-state resident in registers, h double-buffered bf16 in global scratch,
// one device-wide generation barrier per timestep.
// ---------------------------------------------------------------------------
__global__ __launch_bounds__(256)
void lstm_persistent(const float* __restrict__ y_hist,  // (B,T)
                     const float* __restrict__ w_ih,    // (4H)
                     const float* __restrict__ W_hh,    // (4H,H) row-major
                     const float* __restrict__ b_ih,    // (4H)
                     const float* __restrict__ b_hh,    // (4H)
                     const float* __restrict__ c0,      // (B,H)
                     __bf16* __restrict__ hb0,
                     __bf16* __restrict__ hb1,
                     unsigned* __restrict__ bar)        // [0]=count, [1]=generation
{
    extern __shared__ __bf16 ldsW[];   // [4 gates][KIT][32 lanes][16 bf16] = 128 KB

    const int tid  = threadIdx.x;
    const int lane = tid & 31;
    const int wave = tid >> 5;         // 0..7 -> M-tile (16 batch rows each)
    const int j0   = blockIdx.x * NCOL;

    // ---- one-time: swizzle this WG's W_hh slice into LDS as B fragments ----
    // B[k][n] = W_hh[gate*H + j0 + n][k]; B-frag: lane = n (+16 dup),
    // lane<16 holds K = kb+0..15 with kb=0, lane>=16 with kb=16.
    for (int idx = tid; idx < 4 * KIT * 32 * 16; idx += 256) {
        int p   = idx & 15;
        int ln  = (idx >> 4) & 31;
        int kit = (idx >> 9) & 31;
        int nt  = idx >> 14;
        int n   = ln & 15;
        int kb  = (ln < 16) ? 0 : 16;
        int k   = kit * 32 + kb + p;
        long r  = (long)nt * Hsz + j0 + n;
        ldsW[idx] = (__bf16)W_hh[r * Hsz + k];
    }
    __syncthreads();

    // ---- per-thread invariants ----
    const int nloc = lane & 15;        // column within 16-wide tile
    const int j    = j0 + nloc;        // global hidden column
    float wv[4], bv[4];                // per-gate rank-1 weight and fused bias
#pragma unroll
    for (int nt = 0; nt < 4; ++nt) {
        int r  = nt * Hsz + j;
        wv[nt] = w_ih[r];
        bv[nt] = b_ih[r] + b_hh[r];
    }

    // C/D accumulator layout: VGPR r -> M = r (lanes 0-15) or r+8 (lanes 16-31)
    const int  mlo   = (lane < 16) ? 0 : 8;
    const int  brow0 = wave * 16 + mlo;

    // persistent cell state in registers, same layout as accumulators
    v8f c;
#pragma unroll
    for (int r = 0; r < 8; ++r) c[r] = c0[(long)(brow0 + r) * Hsz + j];

    // A-fragment addressing (contiguous k order, same as the LDS B fragments):
    // lane<16: K = 0..15 ; lane>=16: K = 16..31
    const int  aKb   = (lane < 16) ? 0 : 16;
    const long abase = (long)(wave * 16 + (lane & 15)) * Hsz + aKb;

    const v8f zero = {0.f, 0.f, 0.f, 0.f, 0.f, 0.f, 0.f, 0.f};

    for (int t = 0; t < Tsz; ++t) {
        const __bf16* hin  = (t & 1) ? hb1 : hb0;
        __bf16*       hout = (t & 1) ? hb0 : hb1;

        v8f acc[4];
#pragma unroll
        for (int nt = 0; nt < 4; ++nt) acc[nt] = zero;

        for (int kit = 0; kit < KIT; ++kit) {
            const __bf16* ap = hin + abase + kit * 32;
            v8bf alo = *(const v8bf*)ap;
            v8bf ahi = *(const v8bf*)(ap + 8);
            v16bf a  = __builtin_shufflevector(alo, ahi, 0, 1, 2, 3, 4, 5, 6, 7,
                                               8, 9, 10, 11, 12, 13, 14, 15);
#pragma unroll
            for (int nt = 0; nt < 4; ++nt) {
                v16bf b = *(const v16bf*)&ldsW[((((nt * KIT + kit) << 5) + lane) << 4)];
                acc[nt] = __builtin_amdgcn_wmma_f32_16x16x32_bf16(
                    false, a, false, b, (short)0, acc[nt], false, false);
            }
        }

        // elementwise LSTM cell update: acc[0..3] are i,f,g,o aligned per thread
#pragma unroll
        for (int r = 0; r < 8; ++r) {
            float x  = y_hist[(long)(brow0 + r) * Tsz + t];
            float gi = sigmoidf_(acc[0][r] + x * wv[0] + bv[0]);
            float gf = sigmoidf_(acc[1][r] + x * wv[1] + bv[1]);
            float gg = tanhf(acc[2][r] + x * wv[2] + bv[2]);
            float go = sigmoidf_(acc[3][r] + x * wv[3] + bv[3]);
            float cn = gf * c[r] + gi * gg;
            c[r] = cn;
            float hn = go * tanhf(cn);
            hout[(long)(brow0 + r) * Hsz + j] = (__bf16)hn;
        }

        // ---- device-wide barrier, generation t+1 ----
        __syncthreads();
        if (tid == 0) {
            unsigned prev = __hip_atomic_fetch_add(&bar[0], 1u, __ATOMIC_ACQ_REL,
                                                   __HIP_MEMORY_SCOPE_AGENT);
            if (prev == (unsigned)(t + 1) * NWG - 1u) {
                __hip_atomic_store(&bar[1], (unsigned)(t + 1), __ATOMIC_RELEASE,
                                   __HIP_MEMORY_SCOPE_AGENT);
            } else {
                while (__hip_atomic_load(&bar[1], __ATOMIC_ACQUIRE,
                                         __HIP_MEMORY_SCOPE_AGENT) < (unsigned)(t + 1))
                    __builtin_amdgcn_s_sleep(8);
            }
        }
        __syncthreads();
    }
}

// ---------------------------------------------------------------------------
// Final FC: out(128x128) = h_T(128x1024) @ fc_W^T + fc_b, WMMA bf16.
// grid 8 (M-tiles) x 8 waves (N-tiles); one 16x16 tile per wave.
// ---------------------------------------------------------------------------
__global__ __launch_bounds__(256)
void fc_kernel(const __bf16* __restrict__ hT,   // (B,H) bf16 final hidden
               const float*  __restrict__ fcW,  // (O,H)
               const float*  __restrict__ fcb,  // (O)
               float*        __restrict__ out)  // (B,O)
{
    const int tid  = threadIdx.x;
    const int lane = tid & 31;
    const int wave = tid >> 5;
    const int mt   = blockIdx.x;     // 0..7
    const int nloc = lane & 15;
    const int ncol = wave * 16 + nloc;

    const int  aKb   = (lane < 16) ? 0 : 16;
    const long abase = (long)(mt * 16 + (lane & 15)) * Hsz + aKb;
    const int  bKb   = (lane < 16) ? 0 : 16;

    const v8f zero = {0.f, 0.f, 0.f, 0.f, 0.f, 0.f, 0.f, 0.f};
    v8f acc = zero;

    for (int kit = 0; kit < KIT; ++kit) {
        const __bf16* ap = hT + abase + kit * 32;
        v8bf alo = *(const v8bf*)ap;
        v8bf ahi = *(const v8bf*)(ap + 8);
        v16bf a  = __builtin_shufflevector(alo, ahi, 0, 1, 2, 3, 4, 5, 6, 7,
                                           8, 9, 10, 11, 12, 13, 14, 15);
        const float* wp = fcW + (long)ncol * Hsz + kit * 32 + bKb;
        v8f flo = *(const v8f*)wp;
        v8f fhi = *(const v8f*)(wp + 8);
        v16bf b;
#pragma unroll
        for (int p = 0; p < 8; ++p) {
            b[p]     = (__bf16)flo[p];
            b[p + 8] = (__bf16)fhi[p];
        }
        acc = __builtin_amdgcn_wmma_f32_16x16x32_bf16(
            false, a, false, b, (short)0, acc, false, false);
    }

    const int  mlo  = (lane < 16) ? 0 : 8;
    const float bias = fcb[ncol];
#pragma unroll
    for (int r = 0; r < 8; ++r)
        out[(long)(mt * 16 + mlo + r) * Osz + ncol] = acc[r] + bias;
}

// ---------------------------------------------------------------------------
extern "C" void kernel_launch(void* const* d_in, const int* in_sizes, int n_in,
                              void* d_out, int out_size, void* d_ws, size_t ws_size,
                              hipStream_t stream) {
    const float* y_hist = (const float*)d_in[0];
    const float* W_ih   = (const float*)d_in[1];
    const float* W_hh   = (const float*)d_in[2];
    const float* b_ih   = (const float*)d_in[3];
    const float* b_hh   = (const float*)d_in[4];
    const float* fc_W   = (const float*)d_in[5];
    const float* fc_b   = (const float*)d_in[6];
    const float* h0     = (const float*)d_in[7];
    const float* c0     = (const float*)d_in[8];
    float* out = (float*)d_out;

    char*     ws  = (char*)d_ws;
    unsigned* bar = (unsigned*)ws;                               // 8 B used
    __bf16*   hb0 = (__bf16*)(ws + 1024);                        // 256 KB
    __bf16*   hb1 = (__bf16*)(ws + 1024 + (size_t)Bsz * Hsz * 2);// 256 KB

    lstm_init<<<dim3((Bsz * Hsz + 255) / 256), dim3(256), 0, stream>>>(h0, hb0, bar);

    const size_t ldsBytes = (size_t)4 * KIT * 32 * 16 * sizeof(__bf16); // 128 KB
    lstm_persistent<<<dim3(NWG), dim3(256), ldsBytes, stream>>>(
        y_hist, W_ih, W_hh, b_ih, b_hh, c0, hb0, hb1, bar);

    // h for t=511 was written to buffer 0
    fc_kernel<<<dim3(8), dim3(256), 0, stream>>>(hb0, fc_W, fc_b, out);
}
